// MultiHeadAttention_42786464202975
// MI455X (gfx1250) — matrix operand order repaired and checked by reading, not compiled, not run on hardware
//
#include <hip/hip_runtime.h>
#include <hip/hip_bf16.h>
#include <hip/hip_fp16.h>

typedef __attribute__((ext_vector_type(16))) _Float16 v16h;
typedef __attribute__((ext_vector_type(8)))  _Float16 v8h;
typedef __attribute__((ext_vector_type(8)))  float    v8f;
typedef __attribute__((ext_vector_type(4)))  int      v4i;

#define D_    1024
#define H_    16
#define DH_   64
#define B_    2
#define S_    2048
#define MTOT_ (B_ * S_)   // 4096

#if __has_builtin(__builtin_amdgcn_global_load_async_to_lds_b128)
#define HAS_ASYNC_LDS 1
#else
#define HAS_ASYNC_LDS 0
#endif

static __device__ __forceinline__ void async16(const _Float16* g, _Float16* l) {
#if HAS_ASYNC_LDS
  __builtin_amdgcn_global_load_async_to_lds_b128(
      (__attribute__((address_space(1))) v4i*)(g),
      (__attribute__((address_space(3))) v4i*)(l), 0, 0);
#else
  *reinterpret_cast<v8h*>(l) = *reinterpret_cast<const v8h*>(g);
#endif
}

static __device__ __forceinline__ void wait_async_zero() {
#if HAS_ASYNC_LDS
#if __has_builtin(__builtin_amdgcn_s_wait_asynccnt)
  __builtin_amdgcn_s_wait_asynccnt(0);
#else
  asm volatile("s_wait_asynccnt 0x0" ::: "memory");
#endif
#endif
}

static __device__ __forceinline__ v16h load_v16h_pair(const _Float16* p0, const _Float16* p1) {
  union { v16h v; v8h h[2]; } u;
  u.h[0] = *reinterpret_cast<const v8h*>(p0);
  u.h[1] = *reinterpret_cast<const v8h*>(p1);
  return u.v;
}

// ---------------- elementwise f32 -> f16 convert ----------------
struct alignas(8) Half4 { _Float16 x, y, z, w; };

__global__ void cvt_f32_f16(const float* __restrict__ src, _Float16* __restrict__ dst, int n4) {
  int i = blockIdx.x * blockDim.x + threadIdx.x;
  if (i < n4) {
    float4 v = reinterpret_cast<const float4*>(src)[i];
    Half4 o;
    o.x = (_Float16)v.x; o.y = (_Float16)v.y; o.z = (_Float16)v.z; o.w = (_Float16)v.w;
    reinterpret_cast<Half4*>(dst)[i] = o;
  }
}

// ---------------- Wt[k][n] = (f16) W[n][k]  (D x D) ----------------
__global__ void transpose_cvt(const float* __restrict__ W, _Float16* __restrict__ Wt) {
  __shared__ float tile[32][33];
  const int k0 = blockIdx.x * 32, n0 = blockIdx.y * 32;
  const int tx = threadIdx.x, ty = threadIdx.y;
#pragma unroll
  for (int i = 0; i < 32; i += 8)
    tile[ty + i][tx] = W[(size_t)(n0 + ty + i) * D_ + k0 + tx];
  __syncthreads();
#pragma unroll
  for (int i = 0; i < 32; i += 8)
    Wt[(size_t)(k0 + ty + i) * D_ + n0 + tx] = (_Float16)tile[tx][ty + i];
}

// ---------------- GEMM: Out = A(MxK,f16) * Bt(KxN,f16) + bias ----------------
// 256 threads = 8 waves; block tile 128(M) x 128(N); K-step 32.
// B tile double-buffered in LDS, staged with async global->LDS copies.
constexpr int OUT_F16_ROW   = 0;  // f16 [M, N] row-major
constexpr int OUT_F16_HEADT = 1;  // f16 [B, H, DH, S] (per-head transposed, for K)
constexpr int OUT_F32_ROW   = 2;  // f32 [M, N] row-major (final output)

template <int MODE>
__global__ __launch_bounds__(256)
void gemm_wmma(const _Float16* __restrict__ A, const _Float16* __restrict__ Bt,
               const float* __restrict__ bias, void* __restrict__ Out) {
  __shared__ _Float16 Bs[2][32 * 136];  // double-buffered 32 K-rows x 128 cols (pad 136)
  const int tid  = threadIdx.x;
  const int wave = tid >> 5;
  const int lane = tid & 31;
  const int lr   = lane & 15;
  const int hs   = lane >> 4;
  const int m0   = blockIdx.x * 128 + wave * 16;
  const int n0   = blockIdx.y * 128;

  v8f zero = {};
  v8f c[8];
#pragma unroll
  for (int i = 0; i < 8; i++) c[i] = zero;

  const int rowB = tid >> 3;            // 0..31
  const int colB = (tid & 7) * 16;      // 0..112

  auto stageB = [&](int buf, int k0) {
    const _Float16* src = Bt + (size_t)(k0 + rowB) * D_ + n0 + colB;
    async16(src,     &Bs[buf][rowB * 136 + colB]);
    async16(src + 8, &Bs[buf][rowB * 136 + colB + 8]);
  };

  stageB(0, 0);
  wait_async_zero();
  __syncthreads();

  for (int k0 = 0; k0 < D_; k0 += 32) {
    const int cur = (k0 >> 5) & 1;
    if (k0 + 32 < D_) stageB(1 - cur, k0 + 32);   // overlap copy with compute

    // A fragment (16x32): lanes 0-15 -> K 0-7 & 16-23 ; lanes 16-31 -> K 8-15 & 24-31
    const _Float16* ap = A + (size_t)(m0 + lr) * D_ + k0 + (hs ? 8 : 0);
    v16h a = load_v16h_pair(ap, ap + 16);

#pragma unroll
    for (int nt = 0; nt < 8; nt++) {
      v16h b;
#pragma unroll
      for (int s = 0; s < 16; s++)
        b[s] = Bs[cur][((s & 7) + 8 * hs + (s >> 3) * 16) * 136 + nt * 16 + lr];
      c[nt] = __builtin_amdgcn_wmma_f32_16x16x32_f16(false, a, false, b,
                                                     (short)0, c[nt], false, false);
    }
    wait_async_zero();
    __syncthreads();
  }

#pragma unroll
  for (int nt = 0; nt < 8; nt++) {
    const int   n  = n0 + nt * 16 + lr;
    const float bv = bias[n];
#pragma unroll
    for (int r = 0; r < 8; r++) {
      const int m   = m0 + r + 8 * hs;   // C layout: VGPR r -> M=r (+8 for upper half)
      const float val = c[nt][r] + bv;
      if (MODE == OUT_F16_ROW) {
        ((_Float16*)Out)[(size_t)m * D_ + n] = (_Float16)val;
      } else if (MODE == OUT_F16_HEADT) {
        const int bb = m >> 11, s = m & (S_ - 1);
        const int h = n >> 6, dd = n & 63;
        ((_Float16*)Out)[(((size_t)bb * H_ + h) * DH_ + dd) * S_ + s] = (_Float16)val;
      } else {
        ((float*)Out)[(size_t)m * D_ + n] = val;
      }
    }
  }
}

// ---------------- flash attention ----------------
// One block (8 waves) = one (b, h, 128-query block). K/V chunks (32 keys) are
// staged once per block into double-buffered LDS with async copies and shared
// by all 8 waves (8x less K/V fetch traffic than per-wave streaming).
__global__ __launch_bounds__(256)
void attn_wmma(const _Float16* __restrict__ qh, const _Float16* __restrict__ khT,
               const _Float16* __restrict__ vh, const int* __restrict__ mask,
               _Float16* __restrict__ ctx) {
  __shared__ _Float16 Ks[2][64 * 40];  // dh 0..63 x keys 0..31 (pad stride 40)
  __shared__ _Float16 Vs[2][32 * 72];  // key 0..31 x dd 0..63 (pad stride 72)
  __shared__ _Float16 pb[8 * 512];     // per-wave 16x32 f16 prob staging

  const int tid  = threadIdx.x;
  const int wave = tid >> 5;
  const int lane = tid & 31;
  const int lr   = lane & 15;
  const int hs   = lane >> 4;
  const int bid  = blockIdx.x;              // 0..511
  const int b    = bid >> 8;
  const int h    = (bid >> 4) & 15;
  const int q0   = (bid & 15) * 128;
  const int m0   = q0 + wave * 16;

  _Float16* pw = &pb[wave * 512];

  const size_t kbase = (size_t)(b * H_ + h) * DH_ * S_;
  const _Float16* vb = vh + (size_t)b * S_ * D_ + h * DH_;
  const int* mbase = mask + (size_t)b * S_ * S_;

  // staging decomposition: K chunk = 64 rows x 64B, V chunk = 32 rows x 128B
  const int kdh   = tid >> 2, kpart = (tid & 3) * 8;   // 64 x 4 x (8 halves)
  const int vkey  = tid >> 3, vpart = (tid & 7) * 8;   // 32 x 8 x (8 halves)

  auto stageKV = [&](int buf, int j0) {
    async16(khT + kbase + (size_t)kdh * S_ + j0 + kpart, &Ks[buf][kdh * 40 + kpart]);
    async16(vb + (size_t)(j0 + vkey) * D_ + vpart,       &Vs[buf][vkey * 72 + vpart]);
  };

  // Q fragments for 16x64 (two 16x32 A-frags along dh)
  const _Float16* qp = qh + (size_t)(b * S_ + m0 + lr) * D_ + h * DH_ + (hs ? 8 : 0);
  const v16h qa0 = load_v16h_pair(qp,      qp + 16);
  const v16h qa1 = load_v16h_pair(qp + 32, qp + 48);

  v8f o0 = {}, o1 = {}, o2 = {}, o3 = {};
  float mx[8], lsum[8];
#pragma unroll
  for (int r = 0; r < 8; r++) { mx[r] = -__builtin_inff(); lsum[r] = 0.f; }

  const float cs = 0.125f * 1.44269504088896340736f;  // (1/sqrt(DH)) * log2(e)

  stageKV(0, 0);
  wait_async_zero();
  __syncthreads();

  for (int j0 = 0; j0 < S_; j0 += 32) {
    const int cur = (j0 >> 5) & 1;
    if (j0 + 32 < S_) stageKV(1 - cur, j0 + 32);   // overlap copy with compute

    // K^T fragments from LDS: B column = key (lr), elements = dh in table order
    v16h bkA0, bkB0, bkA1, bkB1;
#pragma unroll
    for (int s = 0; s < 16; s++) {
      const int kk = (s & 7) + 8 * hs + (s >> 3) * 16;
      bkA0[s] = Ks[cur][kk * 40 + lr];
      bkB0[s] = Ks[cur][(kk + 32) * 40 + lr];
      bkA1[s] = Ks[cur][kk * 40 + 16 + lr];
      bkB1[s] = Ks[cur][(kk + 32) * 40 + 16 + lr];
    }

    v8f zero = {};
    v8f s0 = __builtin_amdgcn_wmma_f32_16x16x32_f16(false, qa0, false, bkA0, (short)0, zero, false, false);
    s0     = __builtin_amdgcn_wmma_f32_16x16x32_f16(false, qa1, false, bkB0, (short)0, s0,   false, false);
    v8f s1 = __builtin_amdgcn_wmma_f32_16x16x32_f16(false, qa0, false, bkA1, (short)0, zero, false, false);
    s1     = __builtin_amdgcn_wmma_f32_16x16x32_f16(false, qa1, false, bkB1, (short)0, s1,   false, false);

    // mask + online softmax (rows live across 16-lane halves; reduce with shuffles)
    float p0[8], p1[8], alpha[8];
#pragma unroll
    for (int r = 0; r < 8; r++) {
      const int sq = m0 + r + 8 * hs;
      const int* mp = mbase + (size_t)sq * S_ + j0 + lr;
      float v0 = (mp[0]  != 0) ? s0[r] : -1e30f;
      float v1 = (mp[16] != 0) ? s1[r] : -1e30f;
      float cm = fmaxf(v0, v1);
#pragma unroll
      for (int off = 1; off < 16; off <<= 1) cm = fmaxf(cm, __shfl_xor(cm, off, 32));
      const float mn = fmaxf(mx[r], cm);
      const float al = exp2f((mx[r] - mn) * cs);
      mx[r] = mn;
      const float e0 = exp2f((v0 - mn) * cs);
      const float e1 = exp2f((v1 - mn) * cs);
      float rs = e0 + e1;
#pragma unroll
      for (int off = 1; off < 16; off <<= 1) rs += __shfl_xor(rs, off, 32);
      lsum[r] = lsum[r] * al + rs;
      p0[r] = e0; p1[r] = e1; alpha[r] = al;
    }
#pragma unroll
    for (int r = 0; r < 8; r++) {
      o0[r] *= alpha[r]; o1[r] *= alpha[r]; o2[r] *= alpha[r]; o3[r] *= alpha[r];
    }

    // C-layout -> A-layout conversion of probabilities via per-wave LDS
#pragma unroll
    for (int r = 0; r < 8; r++) {
      const int m = r + 8 * hs;
      pw[m * 32 + lr]      = (_Float16)p0[r];
      pw[m * 32 + 16 + lr] = (_Float16)p1[r];
    }
    const _Float16* pp = pw + lr * 32 + (hs ? 8 : 0);
    v16h pa = load_v16h_pair(pp, pp + 16);

    // V fragments from LDS: B column = dd (lr), elements = key in table order
    v16h bv0, bv1, bv2, bv3;
#pragma unroll
    for (int s = 0; s < 16; s++) {
      const int kk = (s & 7) + 8 * hs + (s >> 3) * 16;
      bv0[s] = Vs[cur][kk * 72 + lr];
      bv1[s] = Vs[cur][kk * 72 + 16 + lr];
      bv2[s] = Vs[cur][kk * 72 + 32 + lr];
      bv3[s] = Vs[cur][kk * 72 + 48 + lr];
    }
    o0 = __builtin_amdgcn_wmma_f32_16x16x32_f16(false, pa, false, bv0, (short)0, o0, false, false);
    o1 = __builtin_amdgcn_wmma_f32_16x16x32_f16(false, pa, false, bv1, (short)0, o1, false, false);
    o2 = __builtin_amdgcn_wmma_f32_16x16x32_f16(false, pa, false, bv2, (short)0, o2, false, false);
    o3 = __builtin_amdgcn_wmma_f32_16x16x32_f16(false, pa, false, bv3, (short)0, o3, false, false);

    wait_async_zero();
    __syncthreads();
  }

  // normalize + store context (f16 row-major [B,S,D])
#pragma unroll
  for (int r = 0; r < 8; r++) {
    const int m = m0 + r + 8 * hs;
    const float inv = 1.0f / lsum[r];
    _Float16* op = ctx + (size_t)(b * S_ + m) * D_ + h * DH_ + lr;
    op[0]  = (_Float16)(o0[r] * inv);
    op[16] = (_Float16)(o1[r] * inv);
    op[32] = (_Float16)(o2[r] * inv);
    op[48] = (_Float16)(o3[r] * inv);
  }
}

extern "C" void kernel_launch(void* const* d_in, const int* in_sizes, int n_in,
                              void* d_out, int out_size, void* d_ws, size_t ws_size,
                              hipStream_t stream) {
  const float* q    = (const float*)d_in[0];
  const float* k    = (const float*)d_in[1];
  const float* v    = (const float*)d_in[2];
  const int*   mask = (const int*)  d_in[3];
  const float* Wq   = (const float*)d_in[4];
  const float* bq   = (const float*)d_in[5];
  const float* Wk   = (const float*)d_in[6];
  const float* bk   = (const float*)d_in[7];
  const float* Wv   = (const float*)d_in[8];
  const float* bv   = (const float*)d_in[9];
  const float* Wo   = (const float*)d_in[10];
  const float* bo   = (const float*)d_in[11];

  const size_t NE = (size_t)MTOT_ * D_;   // 4M elements
  _Float16* xq16 = (_Float16*)d_ws;
  _Float16* xk16 = xq16 + NE;
  _Float16* xv16 = xk16 + NE;
  _Float16* WtQ  = xv16 + NE;
  _Float16* WtK  = WtQ + (size_t)D_ * D_;
  _Float16* WtV  = WtK + (size_t)D_ * D_;
  _Float16* WtO  = WtV + (size_t)D_ * D_;
  _Float16* qh   = WtO + (size_t)D_ * D_;
  _Float16* khT  = qh  + NE;
  _Float16* vh   = khT + NE;
  _Float16* ctx  = xq16;   // reuse: xq16 dead after Q projection

  const int n4 = (int)(NE / 4);
  dim3 cg((n4 + 255) / 256), cb(256);
  cvt_f32_f16<<<cg, cb, 0, stream>>>(q, xq16, n4);
  cvt_f32_f16<<<cg, cb, 0, stream>>>(k, xk16, n4);
  cvt_f32_f16<<<cg, cb, 0, stream>>>(v, xv16, n4);

  dim3 tg(D_ / 32, D_ / 32), tb(32, 8);
  transpose_cvt<<<tg, tb, 0, stream>>>(Wq, WtQ);
  transpose_cvt<<<tg, tb, 0, stream>>>(Wk, WtK);
  transpose_cvt<<<tg, tb, 0, stream>>>(Wv, WtV);
  transpose_cvt<<<tg, tb, 0, stream>>>(Wo, WtO);

  dim3 gg(MTOT_ / 128, D_ / 128), gb(256);
  gemm_wmma<OUT_F16_ROW>  <<<gg, gb, 0, stream>>>(xq16, WtQ, bq, (void*)qh);
  gemm_wmma<OUT_F16_HEADT><<<gg, gb, 0, stream>>>(xk16, WtK, bk, (void*)khT);
  gemm_wmma<OUT_F16_ROW>  <<<gg, gb, 0, stream>>>(xv16, WtV, bv, (void*)vh);

  attn_wmma<<<dim3(512), dim3(256), 0, stream>>>(qh, khT, vh, mask, ctx);

  gemm_wmma<OUT_F32_ROW><<<gg, gb, 0, stream>>>(ctx, WtO, bo, d_out);
}
